// MultiHeadAttention_33191507263558
// MI455X (gfx1250) — matrix operand order repaired and checked by reading, not compiled, not run on hardware
//
#include <hip/hip_runtime.h>
#include <hip/hip_bf16.h>
#include <math.h>

// ---------------------------------------------------------------- constants
#define SEQ   4096
#define EMB   512
#define NHD   8
#define HDD   64
#define BAT   2
#define MROWS (BAT * SEQ)   // 8192

typedef __attribute__((ext_vector_type(4)))  __bf16 bf16x4;
typedef __attribute__((ext_vector_type(8)))  __bf16 bf16x8;
typedef __attribute__((ext_vector_type(16))) __bf16 bf16x16;
typedef __attribute__((ext_vector_type(8)))  float  f32x8;
typedef __attribute__((ext_vector_type(4)))  int    v4i;

#define GLB_AS __attribute__((address_space(1)))
#define LDS_AS __attribute__((address_space(3)))

// ------------------------------------------------- async global->LDS copies
#if defined(__has_builtin)
#if __has_builtin(__builtin_amdgcn_global_load_async_to_lds_b128) && \
    __has_builtin(__builtin_amdgcn_s_wait_asynccnt)
#define HAVE_ASYNC 1
#endif
#endif
#ifndef HAVE_ASYNC
#define HAVE_ASYNC 0
#endif

__device__ __forceinline__ void cp_g2l_b128(const void* g, void* l) {
#if HAVE_ASYNC
    __builtin_amdgcn_global_load_async_to_lds_b128(
        (GLB_AS v4i*)const_cast<void*>(g), (LDS_AS v4i*)l, 0, 0);
#else
    *(uint4*)l = *(const uint4*)g;
#endif
}
__device__ __forceinline__ void cp_wait_all() {
#if HAVE_ASYNC
    __builtin_amdgcn_s_wait_asynccnt(0);
#endif
}

// ---------------------------------------------------------------- helpers
__device__ __forceinline__ f32x8 wmma_bf16(bf16x16 a, bf16x16 b, f32x8 c) {
    return __builtin_amdgcn_wmma_f32_16x16x32_bf16(false, a, false, b,
                                                   (short)0, c, false, false);
}

// A-matrix fragment (16x32 bf16): 8 contiguous at k0+half*8, 8 at k0+16+half*8.
__device__ __forceinline__ bf16x16 ld_frag_split(const __bf16* row, int k0, int half) {
    const bf16x8 lo = *(const bf16x8*)(row + k0 + half * 8);
    const bf16x8 hi = *(const bf16x8*)(row + k0 + 16 + half * 8);
    bf16x16 r;
#pragma unroll
    for (int i = 0; i < 8; ++i) { r[i] = lo[i]; r[i + 8] = hi[i]; }
    return r;
}

// B-matrix fragment (32x16 bf16): 8 contiguous at k0+half*8, 8 at k0+16+half*8.
__device__ __forceinline__ bf16x16 ld_frag_contig(const __bf16* row, int k0, int half) {
    const bf16x8 lo = *(const bf16x8*)(row + k0 + half * 8);
    const bf16x8 hi = *(const bf16x8*)(row + k0 + 16 + half * 8);
    bf16x16 r;
#pragma unroll
    for (int i = 0; i < 8; ++i) { r[i] = lo[i]; r[i + 8] = hi[i]; }
    return r;
}

__device__ __forceinline__ float redmax16(float v) {
#pragma unroll
    for (int m = 1; m <= 8; m <<= 1) v = fmaxf(v, __shfl_xor(v, m, 32));
    return v;
}
__device__ __forceinline__ float redsum16(float v) {
#pragma unroll
    for (int m = 1; m <= 8; m <<= 1) v += __shfl_xor(v, m, 32);
    return v;
}

__device__ __forceinline__ bf16x4 pack4(float4 f) {
    bf16x4 t;
    t[0] = (__bf16)f.x; t[1] = (__bf16)f.y;
    t[2] = (__bf16)f.z; t[3] = (__bf16)f.w;
    return t;
}

// ------------------------------------------------------- QKV projection GEMM
// Y = X @ W^T, output reordered to bf16 [B, H, S, D].  grid.z selects q/k/v.
__global__ void __launch_bounds__(128)
qkv_proj_kernel(const float* __restrict__ q, const float* __restrict__ k,
                const float* __restrict__ v, const float* __restrict__ Wq,
                const float* __restrict__ Wk, const float* __restrict__ Wv,
                __bf16* __restrict__ Qb, __bf16* __restrict__ Kb,
                __bf16* __restrict__ Vb) {
    const int which = blockIdx.z;
    const float* X = (which == 0) ? q : (which == 1) ? k : v;
    const float* W = (which == 0) ? Wq : (which == 1) ? Wk : Wv;
    __bf16*      O = (which == 0) ? Qb : (which == 1) ? Kb : Vb;

    const int mbase = blockIdx.x * 64;
    const int nbase = blockIdx.y * 64;

    __shared__ __bf16 As[64][40];   // m x k   (pitch 80B, 16B multiple)
    __shared__ __bf16 BsT[64][40];  // n x k

    const int tid = threadIdx.x, wave = tid >> 5, lane = tid & 31;
    const int half = lane >> 4, mm = lane & 15;

    f32x8 acc[4] = {};
    for (int k0 = 0; k0 < EMB; k0 += 32) {
        // batched loads: all b128 global loads in flight before any convert
        float4 xa[4], wb[4];
#pragma unroll
        for (int i = 0; i < 4; ++i) {
            int idx = i * 128 + tid;                   // 0..511 float4s
            int r = idx >> 3, c = (idx & 7) * 4;
            xa[i] = *(const float4*)&X[(size_t)(mbase + r) * EMB + k0 + c];
            wb[i] = *(const float4*)&W[(size_t)(nbase + r) * EMB + k0 + c];
        }
#pragma unroll
        for (int i = 0; i < 4; ++i) {
            int idx = i * 128 + tid;
            int r = idx >> 3, c = (idx & 7) * 4;
            *(bf16x4*)&As[r][c]  = pack4(xa[i]);
            *(bf16x4*)&BsT[r][c] = pack4(wb[i]);
        }
        __syncthreads();
        bf16x16 a = ld_frag_split(&As[wave * 16 + mm][0], 0, half);
#pragma unroll
        for (int nt = 0; nt < 4; ++nt) {
            bf16x16 b = ld_frag_contig(&BsT[nt * 16 + mm][0], 0, half);
            acc[nt] = wmma_bf16(a, b, acc[nt]);
        }
        __syncthreads();
    }
#pragma unroll
    for (int nt = 0; nt < 4; ++nt)
#pragma unroll
        for (int j = 0; j < 8; ++j) {
            int row = mbase + wave * 16 + j + 8 * half;   // C layout: M = j(+8)
            int col = nbase + nt * 16 + mm;               // N = lane&15
            int b_ = row >> 12, s_ = row & (SEQ - 1);
            int h_ = col >> 6,  d_ = col & 63;
            O[(((size_t)(b_ * NHD + h_) * SEQ + s_) << 6) + d_] = (__bf16)acc[nt][j];
        }
}

// ------------------------------------------------------- causal attention
// One WG = one (b, h, 64-row q tile). 4 waves, each owns a 16-row strip.
// Pass 1: online (m, l).  Pass 2: recompute, write attn = p/l, accumulate PV.
__global__ void __launch_bounds__(128)
attn_kernel(const __bf16* __restrict__ Qb, const __bf16* __restrict__ Kb,
            const __bf16* __restrict__ Vb, float* __restrict__ attn_out,
            __bf16* __restrict__ xh) {
    const int qt = blockIdx.x, h = blockIdx.y, b = blockIdx.z;
    const size_t head = (size_t)(b * NHD + h);
    const __bf16* Qh = Qb + head * SEQ * HDD;
    const __bf16* Kh = Kb + head * SEQ * HDD;
    const __bf16* Vh = Vb + head * SEQ * HDD;
    float* Ah = attn_out + head * (size_t)SEQ * SEQ;

    __shared__ __bf16 Qs[64][72];
    __shared__ __bf16 Ks[64][72];
    __shared__ __bf16 Vt[64][72];        // transposed: [d][kpos]
    __shared__ __bf16 Ps[4][16][72];     // per-wave bf16 P tile (PV A-frags)
    __shared__ float  P32[64][68];       // fp32 P tile -> coalesced attn store

    const int tid = threadIdx.x, wave = tid >> 5, lane = tid & 31;
    const int half = lane >> 4, mm = lane & 15;
    const float SCALE = 0.125f;          // 1/sqrt(64)

    // ---- load Q tile (64x64 bf16) via async copies
#pragma unroll
    for (int i = 0; i < 4; ++i) {
        int idx = i * 128 + tid;                    // 0..511 uint4s
        int r = idx >> 3, cb = (idx & 7) * 8;
        cp_g2l_b128(Qh + (size_t)(qt * 64 + r) * HDD + cb, &Qs[r][cb]);
    }
    cp_wait_all();
    __syncthreads();
    const bf16x16 aq0 = ld_frag_split(&Qs[wave * 16 + mm][0], 0, half);
    const bf16x16 aq1 = ld_frag_split(&Qs[wave * 16 + mm][0], 32, half);

    float mrow[8], lrow[8];
#pragma unroll
    for (int j = 0; j < 8; ++j) { mrow[j] = -1e30f; lrow[j] = 0.0f; }

    // ================= pass 1: row max / row sum (online softmax stats)
    for (int kt = 0; kt <= qt; ++kt) {
#pragma unroll
        for (int i = 0; i < 4; ++i) {
            int idx = i * 128 + tid;
            int r = idx >> 3, cb = (idx & 7) * 8;
            cp_g2l_b128(Kh + (size_t)(kt * 64 + r) * HDD + cb, &Ks[r][cb]);
        }
        if (kt + 1 <= qt)
            __builtin_prefetch(Kh + (size_t)((kt + 1) * 64 + (tid >> 1)) * HDD, 0, 0);
        cp_wait_all();
        __syncthreads();

        float ev[4][8];
#pragma unroll
        for (int nt = 0; nt < 4; ++nt) {
            bf16x16 b0 = ld_frag_contig(&Ks[nt * 16 + mm][0], 0, half);
            bf16x16 b1 = ld_frag_contig(&Ks[nt * 16 + mm][0], 32, half);
            f32x8 c = {};
            c = wmma_bf16(aq0, b0, c);
            c = wmma_bf16(aq1, b1, c);
            const int kcol = kt * 64 + nt * 16 + mm;
#pragma unroll
            for (int j = 0; j < 8; ++j) {
                int qrow = qt * 64 + wave * 16 + j + 8 * half;
                float x = c[j] * SCALE;
                if (kcol > qrow) x = -1e30f;     // causal mask
                ev[nt][j] = x;
            }
        }
#pragma unroll
        for (int j = 0; j < 8; ++j) {
            float tm = ev[0][j];
#pragma unroll
            for (int nt = 1; nt < 4; ++nt) tm = fmaxf(tm, ev[nt][j]);
            tm = redmax16(tm);
            float mnew = fmaxf(mrow[j], tm);
            float s = 0.0f;
#pragma unroll
            for (int nt = 0; nt < 4; ++nt) s += __expf(ev[nt][j] - mnew);
            s = redsum16(s);
            lrow[j] = lrow[j] * __expf(mrow[j] - mnew) + s;
            mrow[j] = mnew;
        }
        __syncthreads();
    }

    float invl[8];
#pragma unroll
    for (int j = 0; j < 8; ++j) invl[j] = 1.0f / lrow[j];

    // ================= pass 2: write attn, accumulate P @ V
    f32x8 acc[4] = {};
    for (int kt = 0; kt <= qt; ++kt) {
        // K tile async; V tile loaded as vectors then transposed into LDS
#pragma unroll
        for (int i = 0; i < 4; ++i) {
            int idx = i * 128 + tid;
            int r = idx >> 3, cb = (idx & 7) * 8;
            cp_g2l_b128(Kh + (size_t)(kt * 64 + r) * HDD + cb, &Ks[r][cb]);
        }
        bf16x8 vrow[4];
#pragma unroll
        for (int i = 0; i < 4; ++i) {
            int idx = i * 128 + tid;
            int kp = idx >> 3, cb = (idx & 7) * 8;
            vrow[i] = *(const bf16x8*)(Vh + (size_t)(kt * 64 + kp) * HDD + cb);
        }
#pragma unroll
        for (int i = 0; i < 4; ++i) {
            int idx = i * 128 + tid;
            int kp = idx >> 3, cb = (idx & 7) * 8;
#pragma unroll
            for (int e = 0; e < 8; ++e) Vt[cb + e][kp] = vrow[i][e];
        }
        if (kt + 1 <= qt)
            __builtin_prefetch(Vh + (size_t)((kt + 1) * 64 + (tid >> 1)) * HDD, 0, 0);
        cp_wait_all();
        __syncthreads();

#pragma unroll
        for (int nt = 0; nt < 4; ++nt) {
            bf16x16 b0 = ld_frag_contig(&Ks[nt * 16 + mm][0], 0, half);
            bf16x16 b1 = ld_frag_contig(&Ks[nt * 16 + mm][0], 32, half);
            f32x8 c = {};
            c = wmma_bf16(aq0, b0, c);
            c = wmma_bf16(aq1, b1, c);
            const int kcol = kt * 64 + nt * 16 + mm;
#pragma unroll
            for (int j = 0; j < 8; ++j) {
                int r = wave * 16 + j + 8 * half;
                int qrow = qt * 64 + r;
                float p = (kcol > qrow)
                              ? 0.0f
                              : __expf(c[j] * SCALE - mrow[j]) * invl[j];
                P32[r][nt * 16 + mm] = p;                 // staged fp32 tile
                Ps[wave][r - wave * 16][nt * 16 + mm] = (__bf16)p;
            }
        }
        // PV: x_tile(16x64) += P(16x64) @ V(64x64)  (wave-local LDS)
        bf16x16 pa0 = ld_frag_split(&Ps[wave][mm][0], 0, half);
        bf16x16 pa1 = ld_frag_split(&Ps[wave][mm][0], 32, half);
#pragma unroll
        for (int nt = 0; nt < 4; ++nt) {
            bf16x16 vb0 = ld_frag_contig(&Vt[nt * 16 + mm][0], 0, half);
            bf16x16 vb1 = ld_frag_contig(&Vt[nt * 16 + mm][0], 32, half);
            acc[nt] = wmma_bf16(pa0, vb0, acc[nt]);
            acc[nt] = wmma_bf16(pa1, vb1, acc[nt]);
        }
        __syncthreads();                      // all P32 rows produced

        // cooperative, fully coalesced attn tile store (b128 per lane)
#pragma unroll
        for (int i = 0; i < 8; ++i) {
            int idx = i * 128 + tid;          // 0..1023 float4s
            int r = idx >> 4, c4 = (idx & 15) * 4;
            *(float4*)&Ah[(size_t)(qt * 64 + r) * SEQ + kt * 64 + c4] =
                *(const float4*)&P32[r][c4];
        }
        __syncthreads();                      // before Ks/Vt overwrite
    }

    // ---- per-head context out, re-interleaved to [B, S, H*D] (bf16)
#pragma unroll
    for (int nt = 0; nt < 4; ++nt)
#pragma unroll
        for (int j = 0; j < 8; ++j) {
            int row = qt * 64 + wave * 16 + j + 8 * half;
            int d = nt * 16 + mm;
            xh[((size_t)(b * SEQ + row) * EMB) + h * HDD + d] = (__bf16)acc[nt][j];
        }

    // ---- zero-fill the strictly-masked region of attn (no compute needed)
    const int rem = SEQ - (qt + 1) * 64;
    if (rem > 0) {
        const int rem4 = rem >> 2;
        for (int idx = tid; idx < 64 * rem4; idx += 128) {
            int r = idx / rem4, c = idx % rem4;
            *(float4*)&Ah[(size_t)(qt * 64 + r) * SEQ + (qt + 1) * 64 + c * 4] =
                make_float4(0.f, 0.f, 0.f, 0.f);
        }
    }
}

// ------------------------------------------------------- output projection
// x = xh @ Wo^T + bo   (xh already bf16 in workspace), fp32 out.
__global__ void __launch_bounds__(128)
out_proj_kernel(const __bf16* __restrict__ xh, const float* __restrict__ Wo,
                const float* __restrict__ bo, float* __restrict__ xout) {
    const int mbase = blockIdx.x * 64;
    const int nbase = blockIdx.y * 64;

    __shared__ __bf16 As[64][40];
    __shared__ __bf16 BsT[64][40];

    const int tid = threadIdx.x, wave = tid >> 5, lane = tid & 31;
    const int half = lane >> 4, mm = lane & 15;

    f32x8 acc[4] = {};
    for (int k0 = 0; k0 < EMB; k0 += 32) {
        // A is already bf16: async copy straight to LDS (16B granules)
#pragma unroll
        for (int i = 0; i < 2; ++i) {
            int idx = i * 128 + tid;                   // 0..255 uint4s
            int r = idx >> 2, cb = (idx & 3) * 8;
            cp_g2l_b128(xh + (size_t)(mbase + r) * EMB + k0 + cb, &As[r][cb]);
        }
        // W fp32 -> batched b128 loads, then packed convert
        float4 wb[4];
#pragma unroll
        for (int i = 0; i < 4; ++i) {
            int idx = i * 128 + tid;
            int r = idx >> 3, c = (idx & 7) * 4;
            wb[i] = *(const float4*)&Wo[(size_t)(nbase + r) * EMB + k0 + c];
        }
#pragma unroll
        for (int i = 0; i < 4; ++i) {
            int idx = i * 128 + tid;
            int r = idx >> 3, c = (idx & 7) * 4;
            *(bf16x4*)&BsT[r][c] = pack4(wb[i]);
        }
        cp_wait_all();
        __syncthreads();
        bf16x16 a = ld_frag_split(&As[wave * 16 + mm][0], 0, half);
#pragma unroll
        for (int nt = 0; nt < 4; ++nt) {
            bf16x16 b = ld_frag_contig(&BsT[nt * 16 + mm][0], 0, half);
            acc[nt] = wmma_bf16(a, b, acc[nt]);
        }
        __syncthreads();
    }
#pragma unroll
    for (int nt = 0; nt < 4; ++nt)
#pragma unroll
        for (int j = 0; j < 8; ++j) {
            int row = mbase + wave * 16 + j + 8 * half;
            int col = nbase + nt * 16 + mm;
            xout[(size_t)row * EMB + col] = acc[nt][j] + bo[col];
        }
}

// ---------------------------------------------------------------- launcher
extern "C" void kernel_launch(void* const* d_in, const int* in_sizes, int n_in,
                              void* d_out, int out_size, void* d_ws, size_t ws_size,
                              hipStream_t stream) {
    const float* q  = (const float*)d_in[0];
    const float* k  = (const float*)d_in[1];
    const float* v  = (const float*)d_in[2];
    const float* Wq = (const float*)d_in[3];
    const float* Wk = (const float*)d_in[4];
    const float* Wv = (const float*)d_in[5];
    const float* Wo = (const float*)d_in[6];
    const float* bo = (const float*)d_in[7];

    // workspace carve-up: bf16 Q,K,V per head + bf16 context = 32 MB
    const size_t perMat = (size_t)BAT * NHD * SEQ * HDD;   // 4M elements
    char* w = (char*)d_ws;
    __bf16* Qb = (__bf16*)w;
    __bf16* Kb = Qb + perMat;
    __bf16* Vb = Kb + perMat;
    __bf16* xh = Vb + perMat;                              // [B, S, E] bf16

    float* xout = (float*)d_out;                           // (B, S, E) fp32
    float* attn = xout + (size_t)BAT * SEQ * EMB;          // (B, H, S, S) fp32

    qkv_proj_kernel<<<dim3(MROWS / 64, EMB / 64, 3), 128, 0, stream>>>(
        q, k, v, Wq, Wk, Wv, Qb, Kb, Vb);
    attn_kernel<<<dim3(SEQ / 64, NHD, BAT), 128, 0, stream>>>(
        Qb, Kb, Vb, attn, xh);
    out_proj_kernel<<<dim3(MROWS / 64, EMB / 64), 128, 0, stream>>>(
        xh, Wo, bo, xout);
}
